// ProjectedConjugatedCSPNet_89464168776419
// MI455X (gfx1250) — matrix operand order repaired and checked by reading, not compiled, not run on hardware
//
#include <hip/hip_runtime.h>
#include <hip/hip_bf16.h>

typedef __bf16 v16bf __attribute__((ext_vector_type(16)));
typedef float  v8f   __attribute__((ext_vector_type(8)));

#define NN 100000
#define GG 5000
#define EE 1000000
#define HH 128

union Frag32 { struct { uint4 lo, hi; } q; v16bf v; };

static __device__ __forceinline__ v16bf ld_frag16(const __bf16* base) {
  // elements 0..7 at base[0..7], elements 8..15 at base[16..23]
  Frag32 f;
  f.q.lo = *(const uint4*)(base);
  f.q.hi = *(const uint4*)(base + 16);
  return f.v;
}

static __device__ __forceinline__ v16bf ld_frag_contig(const __bf16* p) {
  Frag32 f;
  f.q.lo = ((const uint4*)p)[0];
  f.q.hi = ((const uint4*)p)[1];
  return f.v;
}

static __device__ __forceinline__ float silu_f(float x) {
  return x / (1.0f + __expf(-x));
}

static __device__ __forceinline__ v8f wmma_bf16(v16bf a, v16bf b, v8f c) {
  return __builtin_amdgcn_wmma_f32_16x16x32_bf16(false, a, false, b, (short)0, c,
                                                 false, false);
}

// ---------------- prep: L @ L^T per graph -> bf16 [G][9] ----------------
__global__ void k_ltl(const float* __restrict__ lat, __bf16* __restrict__ ltl) {
  int g = blockIdx.x * blockDim.x + threadIdx.x;
  if (g >= GG) return;
  const float* L = lat + g * 9;
#pragma unroll
  for (int i = 0; i < 3; ++i)
#pragma unroll
    for (int j = 0; j < 3; ++j) {
      float s = L[i*3+0]*L[j*3+0] + L[i*3+1]*L[j*3+1] + L[i*3+2]*L[j*3+2];
      ltl[g * 9 + i * 3 + j] = (__bf16)s;
    }
}

// ---------------- prep: layernorm -> bf16 h, one wave per node ----------------
__global__ __launch_bounds__(256) void k_ln(const float* __restrict__ x,
                                            const float* __restrict__ gw,
                                            const float* __restrict__ bw,
                                            __bf16* __restrict__ hout) {
  int node = blockIdx.x * 8 + (threadIdx.x >> 5);
  if (node >= NN) return;
  int l = threadIdx.x & 31;
  float4 v = *(const float4*)(x + (long)node * HH + l * 4);
  float s = v.x + v.y + v.z + v.w;
#pragma unroll
  for (int o = 16; o > 0; o >>= 1) s += __shfl_xor(s, o);
  float mu = s * (1.0f / 128.0f);
  float dx = v.x - mu, dy = v.y - mu, dz = v.z - mu, dw = v.w - mu;
  float q = dx*dx + dy*dy + dz*dz + dw*dw;
#pragma unroll
  for (int o = 16; o > 0; o >>= 1) q += __shfl_xor(q, o);
  float rstd = rsqrtf(q * (1.0f / 128.0f) + 1e-5f);
  float4 g4 = *(const float4*)(gw + l * 4);
  float4 b4 = *(const float4*)(bw + l * 4);
  union { __bf16 h[4]; uint2 u; } pk;
  pk.h[0] = (__bf16)(dx * rstd * g4.x + b4.x);
  pk.h[1] = (__bf16)(dy * rstd * g4.y + b4.y);
  pk.h[2] = (__bf16)(dz * rstd * g4.z + b4.z);
  pk.h[3] = (__bf16)(dw * rstd * g4.w + b4.w);
  *(uint2*)(hout + (long)node * HH + l * 4) = pk.u;
}

// -------- prep: repack f32 [Kreal][128] weight -> fragment-native bf16 --------
// dst layout: [(kb*8+nt)*32 + lane][16] contiguous per lane (two b128 loads).
// B-fragment element idx -> K = kb*32 + 8*(lane>>4) + (idx&7) + 16*(idx>>3),
// n = nt*16 + (lane&15).  K >= Kreal padded with zeros.
__global__ void k_prep(const float* __restrict__ src, __bf16* __restrict__ dst,
                       int Kreal, int KB) {
  int tid = blockIdx.x * blockDim.x + threadIdx.x;
  if (tid >= KB * 256) return;
  int lane = tid & 31;
  int nt = (tid >> 5) & 7;
  int kb = tid >> 8;
  int hi = lane >> 4, n = nt * 16 + (lane & 15);
#pragma unroll
  for (int idx = 0; idx < 16; ++idx) {
    int K = kb * 32 + 8 * hi + (idx & 7) + 16 * (idx >> 3);
    float v = (K < Kreal) ? src[(long)K * HH + n] : 0.0f;
    dst[(long)tid * 16 + idx] = (__bf16)v;
  }
}

// ---------------- prep: in-degree counts ----------------
__global__ void k_count(const int* __restrict__ idx0, float* __restrict__ cnt) {
  int e = blockIdx.x * blockDim.x + threadIdx.x;
  if (e < EE) atomicAdd(cnt + idx0[e], 1.0f);
}

// ---------------- edge model: fused gather + MLP(268->128->128) + scatter ----
__global__ __launch_bounds__(128) void k_edge(
    const __bf16* __restrict__ h, const __bf16* __restrict__ ltl,
    const int* __restrict__ idx0, const int* __restrict__ idx1,
    const int* __restrict__ e2g, const float* __restrict__ frac,
    const __bf16* __restrict__ w1f, const float* __restrict__ b1,
    const __bf16* __restrict__ w2f, const float* __restrict__ b2,
    float* __restrict__ agg) {
  __shared__ __bf16 sA[64 * 288];   // 64 edges x 288 (K padded) bf16
  __shared__ __bf16 sS[64 * 128];   // layer-1 activations (per-wave restage)
  const int t = threadIdx.x;
  const long eb = (long)blockIdx.x * 64;

  { // cooperative A-tile build: 2 threads per edge
    const int s = t >> 1, half = t & 1;
    const long e = eb + s;
    const int i = idx0[e], j = idx1[e];
    const uint4* si = (const uint4*)(h + (long)i * HH + half * 64);
    const uint4* sj = (const uint4*)(h + (long)j * HH + half * 64);
    uint4* di = (uint4*)(sA + s * 288 + half * 64);
    uint4* dj = (uint4*)(sA + s * 288 + 128 + half * 64);
#pragma unroll
    for (int q = 0; q < 8; ++q) { di[q] = si[q]; dj[q] = sj[q]; }
    if (half == 0) {
      const int g = e2g[e];
#pragma unroll
      for (int q = 0; q < 9; ++q) sA[s * 288 + 256 + q] = ltl[g * 9 + q];
#pragma unroll
      for (int q = 0; q < 3; ++q) sA[s * 288 + 265 + q] = (__bf16)frac[e * 3 + q];
#pragma unroll
      for (int q = 268; q < 288; ++q) sA[s * 288 + q] = (__bf16)0.0f;
    }
  }
  __syncthreads();

  const int w = t >> 5, l = t & 31;
  const int hi = l >> 4, nl = l & 15;
  const int mrow = w * 16;          // this wave's 16 edges within the tile

  float bias1[8], bias2[8];
#pragma unroll
  for (int nt = 0; nt < 8; ++nt) { bias1[nt] = b1[nt*16+nl]; bias2[nt] = b2[nt*16+nl]; }

  const v8f vz = {0.f,0.f,0.f,0.f,0.f,0.f,0.f,0.f};
  v8f acc[8];
#pragma unroll
  for (int nt = 0; nt < 8; ++nt) acc[nt] = vz;

  const __bf16* arow = sA + (mrow + nl) * 288;
#pragma unroll
  for (int kb = 0; kb < 9; ++kb) {
    v16bf a = ld_frag16(arow + kb * 32 + 8 * hi);
#pragma unroll
    for (int nt = 0; nt < 8; ++nt) {
      v16bf b = ld_frag_contig(w1f + (long)((kb * 8 + nt) * 32 + l) * 16);
      acc[nt] = wmma_bf16(a, b, acc[nt]);
    }
  }

  // bias + SiLU, restage as bf16 A-matrix for layer 2 (same-wave LDS, in-order)
#pragma unroll
  for (int nt = 0; nt < 8; ++nt)
#pragma unroll
    for (int r = 0; r < 8; ++r)
      sS[(mrow + r + 8 * hi) * HH + nt * 16 + nl] =
          (__bf16)silu_f(acc[nt][r] + bias1[nt]);

  v8f acc2[8];
#pragma unroll
  for (int nt = 0; nt < 8; ++nt) acc2[nt] = vz;
  const __bf16* srow = sS + (mrow + nl) * HH;
#pragma unroll
  for (int kb = 0; kb < 4; ++kb) {
    v16bf a = ld_frag16(srow + kb * 32 + 8 * hi);
#pragma unroll
    for (int nt = 0; nt < 8; ++nt) {
      v16bf b = ld_frag_contig(w2f + (long)((kb * 8 + nt) * 32 + l) * 16);
      acc2[nt] = wmma_bf16(a, b, acc2[nt]);
    }
  }

  // bias + SiLU + scatter-add into agg[edge_index0]
  int dsts[8];
#pragma unroll
  for (int r = 0; r < 8; ++r) dsts[r] = idx0[eb + mrow + 8 * hi + r];
#pragma unroll
  for (int nt = 0; nt < 8; ++nt)
#pragma unroll
    for (int r = 0; r < 8; ++r) {
      float v = silu_f(acc2[nt][r] + bias2[nt]);
      atomicAdd(agg + (long)dsts[r] * HH + nt * 16 + nl, v);
    }
}

// ---------------- node model: [h | agg/cnt] -> MLP(256->128->128) + residual --
__global__ __launch_bounds__(128) void k_node(
    const __bf16* __restrict__ h, const float* __restrict__ agg,
    const float* __restrict__ cnt,
    const __bf16* __restrict__ w1f, const float* __restrict__ b1,
    const __bf16* __restrict__ w2f, const float* __restrict__ b2,
    const float* __restrict__ xin, float* __restrict__ out) {
  __shared__ __bf16 sA[64 * 256];
  __shared__ __bf16 sS[64 * 128];
  const int t = threadIdx.x;
  const long nb = (long)blockIdx.x * 64;

  { // build: [h (128) | agg/max(cnt,1) (128)]
    const int s = t >> 1, half = t & 1;
    const long nd = nb + s;
    if (nd < NN) {
      const uint4* sh = (const uint4*)(h + nd * HH + half * 64);
      uint4* dh = (uint4*)(sA + s * 256 + half * 64);
#pragma unroll
      for (int q = 0; q < 8; ++q) dh[q] = sh[q];
      float rc = 1.0f / fmaxf(cnt[nd], 1.0f);
      const float* ar = agg + nd * HH + half * 64;
      __bf16* dr = sA + s * 256 + 128 + half * 64;
#pragma unroll 8
      for (int q = 0; q < 64; ++q) dr[q] = (__bf16)(ar[q] * rc);
    }
  }
  __syncthreads();

  const int w = t >> 5, l = t & 31;
  const int hi = l >> 4, nl = l & 15;
  const int mrow = w * 16;
  if (nb + mrow < NN) {     // wave-uniform; N % 16 == 0 so tiles are full
    float bias1[8], bias2[8];
#pragma unroll
    for (int nt = 0; nt < 8; ++nt) { bias1[nt] = b1[nt*16+nl]; bias2[nt] = b2[nt*16+nl]; }

    const v8f vz = {0.f,0.f,0.f,0.f,0.f,0.f,0.f,0.f};
    v8f acc[8];
#pragma unroll
    for (int nt = 0; nt < 8; ++nt) acc[nt] = vz;

    const __bf16* arow = sA + (mrow + nl) * 256;
#pragma unroll
    for (int kb = 0; kb < 8; ++kb) {
      v16bf a = ld_frag16(arow + kb * 32 + 8 * hi);
#pragma unroll
      for (int nt = 0; nt < 8; ++nt) {
        v16bf b = ld_frag_contig(w1f + (long)((kb * 8 + nt) * 32 + l) * 16);
        acc[nt] = wmma_bf16(a, b, acc[nt]);
      }
    }

#pragma unroll
    for (int nt = 0; nt < 8; ++nt)
#pragma unroll
      for (int r = 0; r < 8; ++r)
        sS[(mrow + r + 8 * hi) * HH + nt * 16 + nl] =
            (__bf16)silu_f(acc[nt][r] + bias1[nt]);

    v8f acc2[8];
#pragma unroll
    for (int nt = 0; nt < 8; ++nt) acc2[nt] = vz;
    const __bf16* srow = sS + (mrow + nl) * HH;
#pragma unroll
    for (int kb = 0; kb < 4; ++kb) {
      v16bf a = ld_frag16(srow + kb * 32 + 8 * hi);
#pragma unroll
      for (int nt = 0; nt < 8; ++nt) {
        v16bf b = ld_frag_contig(w2f + (long)((kb * 8 + nt) * 32 + l) * 16);
        acc2[nt] = wmma_bf16(a, b, acc2[nt]);
      }
    }

#pragma unroll
    for (int nt = 0; nt < 8; ++nt)
#pragma unroll
      for (int r = 0; r < 8; ++r) {
        long nd = nb + mrow + r + 8 * hi;
        int c = nt * 16 + nl;
        float v = silu_f(acc2[nt][r] + bias2[nt]);
        out[nd * HH + c] = xin[nd * HH + c] + v;
      }
  }
}

extern "C" void kernel_launch(void* const* d_in, const int* in_sizes, int n_in,
                              void* d_out, int out_size, void* d_ws, size_t ws_size,
                              hipStream_t stream) {
  const float* node_features = (const float*)d_in[0];
  const float* lattices      = (const float*)d_in[1];
  const int*   edge_index    = (const int*)d_in[2];
  const int*   edge2graph    = (const int*)d_in[3];
  const float* frac_diff     = (const float*)d_in[4];
  const float* W1e = (const float*)d_in[5];
  const float* b1e = (const float*)d_in[6];
  const float* W2e = (const float*)d_in[7];
  const float* b2e = (const float*)d_in[8];
  const float* W1n = (const float*)d_in[9];
  const float* b1n = (const float*)d_in[10];
  const float* W2n = (const float*)d_in[11];
  const float* b2n = (const float*)d_in[12];
  const float* ln_g = (const float*)d_in[13];
  const float* ln_b = (const float*)d_in[14];
  float* out = (float*)d_out;

  size_t off = 0;
  auto carve = [&](size_t bytes) -> void* {
    void* p = (char*)d_ws + off;
    off += (bytes + 255) & ~(size_t)255;
    return p;
  };
  __bf16* h_bf   = (__bf16*)carve((size_t)NN * HH * 2);
  __bf16* ltl_bf = (__bf16*)carve((size_t)GG * 9 * 2);
  __bf16* w1e_f  = (__bf16*)carve((size_t)9 * 256 * 16 * 2);
  __bf16* w2e_f  = (__bf16*)carve((size_t)4 * 256 * 16 * 2);
  __bf16* w1n_f  = (__bf16*)carve((size_t)8 * 256 * 16 * 2);
  __bf16* w2n_f  = (__bf16*)carve((size_t)4 * 256 * 16 * 2);
  float*  agg    = (float*)carve((size_t)NN * HH * 4);
  float*  cnt    = (float*)carve((size_t)NN * 4);
  (void)in_sizes; (void)n_in; (void)out_size; (void)ws_size;

  const int* idx0 = edge_index;        // destination for scatter, source of hi
  const int* idx1 = edge_index + EE;   // source of hj

  hipMemsetAsync(agg, 0, (size_t)NN * HH * 4, stream);
  hipMemsetAsync(cnt, 0, (size_t)NN * 4, stream);

  k_ltl<<<(GG + 127) / 128, 128, 0, stream>>>(lattices, ltl_bf);
  k_ln<<<(NN + 7) / 8, 256, 0, stream>>>(node_features, ln_g, ln_b, h_bf);
  k_prep<<<(9 * 256 + 127) / 128, 128, 0, stream>>>(W1e, w1e_f, 268, 9);
  k_prep<<<(4 * 256 + 127) / 128, 128, 0, stream>>>(W2e, w2e_f, 128, 4);
  k_prep<<<(8 * 256 + 127) / 128, 128, 0, stream>>>(W1n, w1n_f, 256, 8);
  k_prep<<<(4 * 256 + 127) / 128, 128, 0, stream>>>(W2n, w2n_f, 128, 4);
  k_count<<<(EE + 255) / 256, 256, 0, stream>>>(idx0, cnt);

  k_edge<<<EE / 64, 128, 0, stream>>>(h_bf, ltl_bf, idx0, idx1, edge2graph,
                                      frac_diff, w1e_f, b1e, w2e_f, b2e, agg);
  k_node<<<(NN + 63) / 64, 128, 0, stream>>>(h_bf, agg, cnt, w1n_f, b1n,
                                             w2n_f, b2n, node_features, out);
}
